// HOAToBinaural_60069412602253
// MI455X (gfx1250) — matrix operand order repaired and checked by reading, not compiled, not run on hardware
//
#include <hip/hip_runtime.h>

typedef __attribute__((ext_vector_type(16))) _Float16 v16h;
typedef __attribute__((ext_vector_type(2)))  _Float16 v2h;
typedef __attribute__((ext_vector_type(8)))  float    v8f;

#define B_SZ     8
#define NCH      16
#define NSMP     480000
#define NTAP     2048
#define NEARS    2
#define KB_TAPS  256                      // taps per WMMA column group (8 groups x 2 ears)
#define TT       5                        // time tiles per wave (span sharing: 6 spans -> 5 WMMAs)
#define WPB      8                        // waves per block
#define BLOCK_SPAN (WPB * TT * 16)        // 640 output samples per block
#define BLOCKS_PER_B (NSMP / BLOCK_SPAN)  // 750 (exact)
#define HALO     256                      // front halo (255 needed)
#define XS_LEN   (HALO + BLOCK_SPAN)      // 896 staged samples per channel
#define WS_B_BYTES ((size_t)NCH * 8 * 32 * 16 * sizeof(_Float16))   // 128 KB packed filter

__global__ void HOAToBinaural_zero_kernel(float* __restrict__ out, int n) {
    int i = blockIdx.x * blockDim.x + threadIdx.x;
    if (i < n) out[i] = 0.0f;
}

// Pack filter into per-lane WMMA B fragments (f16), tap index REVERSED within
// each K=32 chunk so the A operand is forward-contiguous in time.
// Layout: wsB[(c*8 + kb)*32 + lane][16 halves]  (32B/lane, coalesced b128 reads).
__global__ void HOAToBinaural_packB_kernel(const float* __restrict__ filt,
                                           _Float16* __restrict__ wsB) {
    int t = blockIdx.x * blockDim.x + threadIdx.x;   // 4096 threads: (c, kb, lane)
    int lane  = t & 31;
    int chunk = t >> 5;                               // c*8 + kb
    int kb = chunk & 7;
    int c  = chunk >> 3;
    int hi = lane >> 4;
    int n  = lane & 15;                               // column = g*2 + e
    int e  = n & 1, g = n >> 1;
    const float* f = filt + ((size_t)e * NCH + c) * NTAP + (size_t)g * KB_TAPS;
    _Float16* dst = wsB + ((size_t)chunk * 32 + lane) * 16;
    int k0 = kb * 32;
    #pragma unroll
    for (int i = 0; i < 16; ++i) {
        int K   = (i & 7) + hi * 8 + (i >> 3) * 16;  // B layout: lanes 0-15 hold K=0..7,16..23, lanes 16-31 K=8..15,24..31
        int tap = k0 + 31 - K;        // reversed kappa within chunk
        dst[i] = (_Float16)f[tap];
    }
}

// One block = (batch, 640-sample span). Stage x (f32->f16) into LDS once,
// then 8 waves x 5 tiles of WMMA: P[tau,(g,e)] partial convolutions,
// scatter-added into y at time offset g*256.
template <bool USE_WS>
__global__ __launch_bounds__(256) void HOAToBinaural_wmma_kernel(
    const float* __restrict__ hoa,       // [B, NCH, NSMP]
    const float* __restrict__ filt,      // [NEARS, NCH, NTAP]
    const _Float16* __restrict__ wsB,    // packed B fragments (or null)
    float* __restrict__ out)             // [B, NEARS, NSMP]
{
    __shared__ _Float16 xs[NCH][XS_LEN];             // 28 KB

    const int tid  = threadIdx.x;
    const int lane = tid & 31;
    const int wid  = tid >> 5;
    const int hi   = lane >> 4;
    const int m    = lane & 15;
    const int e    = m & 1;
    const int g    = m >> 1;

    const int b   = blockIdx.x / BLOCKS_PER_B;
    const int t0b = (blockIdx.x % BLOCKS_PER_B) * BLOCK_SPAN;
    const float* xbase = hoa + (size_t)b * NCH * NSMP;

    // ---------------- Stage: x[t0b-256 .. t0b+639] per channel, f32 -> f16 ----------------
    if (t0b >= HALO) {                               // block-uniform: no clamping needed
        for (int c = 0; c < NCH; ++c) {
            const float* xc = xbase + (size_t)c * NSMP + (t0b - HALO);
            v2h* row = (v2h*)&xs[c][0];
            #pragma unroll
            for (int p = tid; p < XS_LEN / 2; p += 256) {
                v2h h; h[0] = (_Float16)xc[2 * p]; h[1] = (_Float16)xc[2 * p + 1];
                row[p] = h;                          // cvt_pk + ds_store_b32
            }
        }
    } else {                                         // first block of each batch: zero-pad t<0
        for (int c = 0; c < NCH; ++c) {
            const float* xc = xbase + (size_t)c * NSMP;
            v2h* row = (v2h*)&xs[c][0];
            #pragma unroll
            for (int p = tid; p < XS_LEN / 2; p += 256) {
                int t = t0b - HALO + 2 * p;
                v2h h;
                h[0] = (_Float16)((t     >= 0) ? xc[t]     : 0.0f);
                h[1] = (_Float16)((t + 1 >= 0) ? xc[t + 1] : 0.0f);
                row[p] = h;
            }
        }
    }
    __syncthreads();

    // ---------------- Compute: 128 chunks x 5 WMMAs per wave ----------------
    const int t0 = t0b + wid * (TT * 16);            // this wave's first output sample
    v8f acc[TT] = {};

    for (int c = 0; c < NCH; ++c) {
        const _Float16* bp = USE_WS ? (wsB + ((size_t)c * 8 * 32 + lane) * 16) : nullptr;
        const float*    fp = USE_WS ? nullptr
                            : (filt + ((size_t)e * NCH + c) * NTAP + (size_t)g * KB_TAPS);
        for (int kb = 0; kb < 8; ++kb) {
            // B fragment: 2 x global_load_b128 from the packed L2-resident filter
            v16h bf;
            if (USE_WS) {
                bf = *(const v16h*)(bp + (size_t)kb * 32 * 16);
            } else {
                #pragma unroll
                for (int i = 0; i < 16; ++i)
                    bf[i] = (_Float16)fp[kb * 32 + 31 - ((i & 7) + hi * 8 + (i >> 3) * 16)];
            }

            // Six 8-half spans from LDS; tile tt uses (span[tt], span[tt+1]).
            // LDS offset of span k, element j: off0 + 16*k + j  (always in [0, 896))
            const int off0 = (wid * (TT * 16) + m + hi * 8) - kb * 32 - 31 + HALO;
            _Float16 sp[TT + 1][8];
            #pragma unroll
            for (int k = 0; k < TT + 1; ++k)
                #pragma unroll
                for (int j = 0; j < 8; ++j)
                    sp[k][j] = xs[c][off0 + 16 * k + j];   // ds_load_u16

            #pragma unroll
            for (int tt = 0; tt < TT; ++tt) {
                v16h a;
                #pragma unroll
                for (int j = 0; j < 8; ++j) { a[j] = sp[tt][j]; a[8 + j] = sp[tt + 1][j]; }
                acc[tt] = __builtin_amdgcn_wmma_f32_16x16x32_f16(
                    false, a, false, bf, (short)0, acc[tt], false, false);
            }
        }
    }

    // ---------------- Scatter: column (g,e), row r+hi*8 -> y[b,e, t0+tt*16+row+g*256] ----------------
    float* ob = out + ((size_t)b * NEARS + e) * NSMP;
    #pragma unroll
    for (int tt = 0; tt < TT; ++tt) {
        #pragma unroll
        for (int r = 0; r < 8; ++r) {
            int t = t0 + tt * 16 + r + hi * 8 + g * KB_TAPS;
            if (t < NSMP) atomicAdd(&ob[t], acc[tt][r]);
        }
    }
}

extern "C" void kernel_launch(void* const* d_in, const int* in_sizes, int n_in,
                              void* d_out, int out_size, void* d_ws, size_t ws_size,
                              hipStream_t stream) {
    const float* hoa  = (const float*)d_in[0];   // [8, 16, 480000] f32
    const float* filt = (const float*)d_in[1];   // [2, 16, 2048]   f32
    float* out = (float*)d_out;                  // [8, 2, 480000]  f32

    HOAToBinaural_zero_kernel<<<(out_size + 255) / 256, 256, 0, stream>>>(out, out_size);

    const int nblocks = B_SZ * BLOCKS_PER_B;     // 6000 blocks x 8 waves

    if (ws_size >= WS_B_BYTES) {
        _Float16* wsB = (_Float16*)d_ws;
        HOAToBinaural_packB_kernel<<<(NCH * 8 * 32) / 256, 256, 0, stream>>>(filt, wsB);
        HOAToBinaural_wmma_kernel<true><<<nblocks, 256, 0, stream>>>(hoa, filt, wsB, out);
    } else {
        HOAToBinaural_wmma_kernel<false><<<nblocks, 256, 0, stream>>>(hoa, filt, nullptr, out);
    }
}
